// CausalCrossConditionalSelfAttention_33724083208556
// MI455X (gfx1250) — matrix operand order repaired and checked by reading, not compiled, not run on hardware
//
#include <hip/hip_runtime.h>
#include <hip/hip_bf16.h>

#define NHEAD 12
#define HD    64
#define TSEQ  2560
#define CDIM  768
#define BATCH 2
#define MODT  256   // t = T / 10
#define MROWS 5120  // B*T

#define USE_ASYNC_LDS 1

typedef __attribute__((ext_vector_type(16))) __bf16 v16bf;
typedef __attribute__((ext_vector_type(8)))  __bf16 v8bf;
typedef __attribute__((ext_vector_type(4)))  __bf16 v4bf;
typedef __attribute__((ext_vector_type(8)))  float  v8f;
typedef __attribute__((ext_vector_type(4)))  float  v4f;

__device__ __forceinline__ v16bf cat8(v8bf lo, v8bf hi) {
  return __builtin_shufflevector(lo, hi, 0,1,2,3,4,5,6,7,8,9,10,11,12,13,14,15);
}
__device__ __forceinline__ v16bf ld2x8(const __bf16* p0, const __bf16* p1) {
  return cat8(*(const v8bf*)p0, *(const v8bf*)p1);
}
__device__ __forceinline__ v8f wmma_bf16(v16bf a, v16bf b, v8f c) {
  // D = A(16x32 bf16) x B(32x16 bf16) + C(16x16 f32)
  return __builtin_amdgcn_wmma_f32_16x16x32_bf16(false, a, false, b, (short)0, c, false, false);
}

// --- CDNA5 async global->LDS copy (16B per lane), tracked by ASYNCcnt -----
// Low 32 bits of a flat __shared__ pointer are the wave-relative LDS offset
// (the shared aperture lives in the high bits), which is exactly what the
// VDST operand of global_load_async_to_lds_* wants.
__device__ __forceinline__ void async_cp16(void* lds, const void* gsrc) {
#if USE_ASYNC_LDS
  unsigned loff = (unsigned)(unsigned long long)lds;
  asm volatile("global_load_async_to_lds_b128 %0, %1, off"
               :: "v"(loff), "v"(gsrc) : "memory");
#else
  *(v8bf*)lds = *(const v8bf*)gsrc;
#endif
}
__device__ __forceinline__ void async_wait0() {
#if USE_ASYNC_LDS
  asm volatile("s_wait_asynccnt 0x0" ::: "memory");
#endif
}

// ---------------------------------------------------------------- convert
__global__ void f32_to_bf16_kernel(const float* __restrict__ src,
                                   __bf16* __restrict__ dst, int n) {
  int i = (blockIdx.x * blockDim.x + threadIdx.x) * 4;
  if (i + 3 < n) {
    v4f v = *(const v4f*)(src + i);
    v4bf o;
    o[0] = (__bf16)v[0]; o[1] = (__bf16)v[1];
    o[2] = (__bf16)v[2]; o[3] = (__bf16)v[3];
    *(v4bf*)(dst + i) = o;
  }
}

// ---------------------------------------------------------------- GEMM
// out[m,n] = sum_k A[m,k] * W[n,k] + bias[n]
// mode 0: write bf16 into [B,H,T,hd] head-major layout (QKV projections)
// mode 1: write f32 row-major MxN (output projection)
__global__ __launch_bounds__(256)
void gemm_bf16_kernel(const __bf16* __restrict__ A, const __bf16* __restrict__ W,
                      const float* __restrict__ bias, int K, int N,
                      int mode, __bf16* __restrict__ dst_bf,
                      float* __restrict__ dst_f) {
  __shared__ __bf16 Asm[128 * 40];
  __shared__ __bf16 Bsm[128 * 40];
  const int tid  = threadIdx.x;
  const int lane = tid & 31;
  const int wave = tid >> 5;
  const int lr   = lane & 15;
  const int hi   = lane >> 4;       // 0: lanes 0-15, 1: lanes 16-31
  const int bm   = blockIdx.y;      // M tile (128 rows)
  const int bn   = blockIdx.x;      // N tile (128 cols)
  const int mw   = (wave >> 1) * 32;
  const int nw   = (wave & 1) * 64;

  v8f zero8 = {0.f, 0.f, 0.f, 0.f, 0.f, 0.f, 0.f, 0.f};
  v8f acc[2][4];
#pragma unroll
  for (int i = 0; i < 2; ++i)
#pragma unroll
    for (int j = 0; j < 4; ++j) acc[i][j] = zero8;

  for (int k0 = 0; k0 < K; k0 += 32) {
    __syncthreads();
#pragma unroll
    for (int cc = tid; cc < 512; cc += 256) {
      int row = cc >> 2, c8 = (cc & 3) * 8;
      const __bf16* ap = A + (size_t)(bm * 128 + row) * K + k0 + c8;
      const __bf16* wp = W + (size_t)(bn * 128 + row) * K + k0 + c8;
      async_cp16(&Asm[row * 40 + c8], ap);   // gfx1250 async global->LDS
      async_cp16(&Bsm[row * 40 + c8], wp);
      if (k0 + 32 < K) {                     // gfx1250 global_prefetch_b8
        __builtin_prefetch(ap + 32, 0, 1);
        __builtin_prefetch(wp + 32, 0, 1);
      }
    }
    async_wait0();
    __syncthreads();

    const int c0 = hi ? 8 : 0;   // A: k chunk base within 32
    const int kb = hi ? 16 : 0;  // unused; B reads the same k chunks (c0) as A
    v16bf af[2], bfg[4];
#pragma unroll
    for (int im = 0; im < 2; ++im) {
      int row = mw + im * 16 + lr;
      af[im] = ld2x8(&Asm[row * 40 + c0], &Asm[row * 40 + c0 + 16]);
    }
#pragma unroll
    for (int in = 0; in < 4; ++in) {
      int col = nw + in * 16 + lr;
      bfg[in] = ld2x8(&Bsm[col * 40 + c0], &Bsm[col * 40 + c0 + 16]);
    }
#pragma unroll
    for (int im = 0; im < 2; ++im)
#pragma unroll
      for (int in = 0; in < 4; ++in)
        acc[im][in] = wmma_bf16(af[im], bfg[in], acc[im][in]);
  }

  // epilogue (b index is block-uniform: 2560/128 = 20 blocks per batch row)
  const int bidx = (bm * 128) / TSEQ;
  const int trow0 = bm * 128 - bidx * TSEQ;
#pragma unroll
  for (int im = 0; im < 2; ++im)
#pragma unroll
    for (int in = 0; in < 4; ++in) {
      int ng = bn * 128 + nw + in * 16 + lr;
      float bv = bias[ng];
      int h = ng >> 6, d = ng & 63;
#pragma unroll
      for (int r = 0; r < 8; ++r) {
        int mloc = mw + im * 16 + (hi ? 8 + r : r);
        float val = acc[im][in][r] + bv;
        if (mode == 0) {
          int tt = trow0 + mloc;
          dst_bf[(((size_t)(bidx * NHEAD + h) * TSEQ + tt) << 6) + d] = (__bf16)val;
        } else {
          dst_f[(size_t)(bm * 128 + mloc) * N + ng] = val;
        }
      }
    }
}

// ---------------------------------------------------------------- attention
// One block = one (b,h) and one 128-row q tile; 8 waves x 16 q rows each.
// Streaming softmax over 32-row K/V tiles; mask: (q%256) >= (k%256).
__global__ __launch_bounds__(256)
void attn_kernel(const __bf16* __restrict__ Qb, const __bf16* __restrict__ Kb,
                 const __bf16* __restrict__ Vb, __bf16* __restrict__ Yb) {
  __shared__ __bf16 Ksm[32 * 72];     // [k][d], row stride 72 (144B)
  __shared__ __bf16 Vtm[64 * 40];     // [d][k], row stride 40 (80B)
  __shared__ __bf16 Psm[8][16 * 40];  // per-wave P scratch [q][k]

  const int tid  = threadIdx.x;
  const int lane = tid & 31;
  const int wave = tid >> 5;
  const int lr   = lane & 15;
  const int hi   = lane >> 4;
  const int bh   = blockIdx.y;            // 0..23
  const int b    = bh / NHEAD;
  const int h    = bh % NHEAD;
  const int qbase = blockIdx.x * 128 + wave * 16;
  const size_t base = (size_t)bh * TSEQ * HD;
  const float scale = 0.125f;             // 1/sqrt(64)

  // Q fragments for this wave's 16 rows (K-dim = 64 -> two A frags)
  const __bf16* qrow = Qb + base + (size_t)(qbase + lr) * HD;
  const int c0 = hi ? 8 : 0;
  v16bf qa0 = ld2x8(qrow + c0, qrow + c0 + 16);
  v16bf qa1 = ld2x8(qrow + 32 + c0, qrow + 32 + c0 + 16);

  v8f zero8 = {0.f, 0.f, 0.f, 0.f, 0.f, 0.f, 0.f, 0.f};
  v8f o[4];
#pragma unroll
  for (int nb = 0; nb < 4; ++nb) o[nb] = zero8;
  float mrow[8], lsum[8];
#pragma unroll
  for (int r = 0; r < 8; ++r) { mrow[r] = -1e30f; lsum[r] = 0.f; }

  for (int kb0 = 0; kb0 < TSEQ; kb0 += 32) {
    __syncthreads();
    {  // stage K tile async (gfx1250), V tile transposed manually
      int kk = tid >> 3, c8 = (tid & 7) * 8;
      const __bf16* ksrc = Kb + base + (size_t)(kb0 + kk) * HD + c8;
      async_cp16(&Ksm[kk * 72 + c8], ksrc);
      v8bf vv = *(const v8bf*)(Vb + base + (size_t)(kb0 + kk) * HD + c8);
#pragma unroll
      for (int j = 0; j < 8; ++j) Vtm[(c8 + j) * 40 + kk] = vv[j];
    }
    async_wait0();
    __syncthreads();

    // S = Q x K^T  (two 16x16 tiles: cols kb0..+15 and kb0+16..+31)
    const int db = hi ? 16 : 0;
    v16bf b00 = ld2x8(&Ksm[(0 + lr) * 72 + 0 + c0],  &Ksm[(0 + lr) * 72 + 0 + c0 + 16]);
    v16bf b01 = ld2x8(&Ksm[(0 + lr) * 72 + 32 + c0], &Ksm[(0 + lr) * 72 + 32 + c0 + 16]);
    v16bf b10 = ld2x8(&Ksm[(16 + lr) * 72 + 0 + c0], &Ksm[(16 + lr) * 72 + 0 + c0 + 16]);
    v16bf b11 = ld2x8(&Ksm[(16 + lr) * 72 + 32 + c0],&Ksm[(16 + lr) * 72 + 32 + c0 + 16]);
    v8f s0 = wmma_bf16(qa1, b01, wmma_bf16(qa0, b00, zero8));
    v8f s1 = wmma_bf16(qa1, b11, wmma_bf16(qa0, b10, zero8));

    // online softmax update
    const int kc0 = (kb0 + lr) % MODT;
    const int kc1 = (kb0 + 16 + lr) % MODT;
    __bf16* pw = &Psm[wave][0];
#pragma unroll
    for (int r = 0; r < 8; ++r) {
      int qr = qbase + (hi ? 8 + r : r);
      int qm = qr % MODT;
      bool ok0 = qm >= kc0, ok1 = qm >= kc1;
      float sv0 = ok0 ? s0[r] * scale : -1e30f;
      float sv1 = ok1 ? s1[r] * scale : -1e30f;
      float mx = fmaxf(sv0, sv1);
#pragma unroll
      for (int off = 1; off < 16; off <<= 1)
        mx = fmaxf(mx, __shfl_xor(mx, off, 16));
      float mnew = fmaxf(mrow[r], mx);
      float alpha = __expf(mrow[r] - mnew);
      float p0 = ok0 ? __expf(sv0 - mnew) : 0.f;
      float p1 = ok1 ? __expf(sv1 - mnew) : 0.f;
      float ps = p0 + p1;
#pragma unroll
      for (int off = 1; off < 16; off <<= 1)
        ps += __shfl_xor(ps, off, 16);
      lsum[r] = lsum[r] * alpha + ps;
      mrow[r] = mnew;
#pragma unroll
      for (int nb = 0; nb < 4; ++nb) o[nb][r] *= alpha;
      int prow = hi ? 8 + r : r;
      pw[prow * 40 + lr]      = (__bf16)p0;
      pw[prow * 40 + 16 + lr] = (__bf16)p1;
    }

    // O += P(16x32) x V(32x64): P via per-wave LDS (same-wave DS ordering)
    v16bf pa = ld2x8(&pw[lr * 40 + c0], &pw[lr * 40 + c0 + 16]);
    const int kkb = hi ? 16 : 0;
#pragma unroll
    for (int nb = 0; nb < 4; ++nb) {
      int d = nb * 16 + lr;
      v16bf vf = ld2x8(&Vtm[d * 40 + c0], &Vtm[d * 40 + c0 + 16]);
      o[nb] = wmma_bf16(pa, vf, o[nb]);
    }
  }

  // normalize and write y in [B,T,C] layout (C index = h*64 + d)
#pragma unroll
  for (int r = 0; r < 8; ++r) lsum[r] = 1.f / lsum[r];
#pragma unroll
  for (int nb = 0; nb < 4; ++nb) {
    int d = nb * 16 + lr;
#pragma unroll
    for (int r = 0; r < 8; ++r) {
      int qr = qbase + (hi ? 8 + r : r);
      Yb[(size_t)(b * TSEQ + qr) * CDIM + h * HD + d] = (__bf16)(o[nb][r] * lsum[r]);
    }
  }
}

// ---------------------------------------------------------------- launch
extern "C" void kernel_launch(void* const* d_in, const int* in_sizes, int n_in,
                              void* d_out, int out_size, void* d_ws, size_t ws_size,
                              hipStream_t stream) {
  (void)in_sizes; (void)n_in; (void)out_size; (void)ws_size;
  const float* x  = (const float*)d_in[0];
  const float* Wk = (const float*)d_in[1];
  const float* bk = (const float*)d_in[2];
  const float* Wq = (const float*)d_in[3];
  const float* bq = (const float*)d_in[4];
  const float* Wv = (const float*)d_in[5];
  const float* bv = (const float*)d_in[6];
  const float* Wp = (const float*)d_in[7];
  const float* bp = (const float*)d_in[8];

  const size_t MK = (size_t)MROWS * CDIM;   // 3,932,160 (== B*H*T*hd)
  const size_t WN = (size_t)CDIM * CDIM;    // 589,824

  char* ws = (char*)d_ws;
  size_t off = 0;
  auto take = [&](size_t elems) {
    void* p = ws + off;
    off = (off + elems * sizeof(__bf16) + 255) & ~(size_t)255;
    return (__bf16*)p;
  };
  __bf16* xbf = take(MK);
  __bf16* wkb = take(WN);
  __bf16* wqb = take(WN);
  __bf16* wvb = take(WN);
  __bf16* wpb = take(WN);
  __bf16* qb  = take(MK);
  __bf16* kb  = take(MK);
  __bf16* vb  = take(MK);
  __bf16* yb  = take(MK);

  f32_to_bf16_kernel<<<(int)((MK / 4 + 255) / 256), 256, 0, stream>>>(x, xbf, (int)MK);
  f32_to_bf16_kernel<<<(int)((WN / 4 + 255) / 256), 256, 0, stream>>>(Wk, wkb, (int)WN);
  f32_to_bf16_kernel<<<(int)((WN / 4 + 255) / 256), 256, 0, stream>>>(Wq, wqb, (int)WN);
  f32_to_bf16_kernel<<<(int)((WN / 4 + 255) / 256), 256, 0, stream>>>(Wv, wvb, (int)WN);
  f32_to_bf16_kernel<<<(int)((WN / 4 + 255) / 256), 256, 0, stream>>>(Wp, wpb, (int)WN);

  dim3 ggrid(CDIM / 128, MROWS / 128);  // (6, 40)
  gemm_bf16_kernel<<<ggrid, 256, 0, stream>>>(xbf, wqb, bq, CDIM, CDIM, 0, qb, nullptr);
  gemm_bf16_kernel<<<ggrid, 256, 0, stream>>>(xbf, wkb, bk, CDIM, CDIM, 0, kb, nullptr);
  gemm_bf16_kernel<<<ggrid, 256, 0, stream>>>(xbf, wvb, bv, CDIM, CDIM, 0, vb, nullptr);

  dim3 agrid(TSEQ / 128, BATCH * NHEAD);  // (20, 24)
  attn_kernel<<<agrid, 256, 0, stream>>>(qb, kb, vb, yb);

  gemm_bf16_kernel<<<ggrid, 256, 0, stream>>>(yb, wpb, bp, CDIM, CDIM, 1, nullptr,
                                              (float*)d_out);
}
